// FSSConv1dCell_73710228734015
// MI455X (gfx1250) — matrix operand order repaired and checked by reading, not compiled, not run on hardware
//
#include <hip/hip_runtime.h>
#include <cstdint>

// ---------------------------------------------------------------------------
// FSSConv1dCell for MI455X (gfx1250):
//  - grouped convs as bf16x3-split WMMA (V_WMMA_F32_16X16X32_BF16)
//  - activation tiles staged global->LDS by the Tensor Data Mover
//    (TENSOR_LOAD_TO_LDS + s_wait_tensorcnt), zero-fill handles edges
//  - channel shuffle folded into stores, fused sigmoid-gate
//  - per-position 256-channel LDS bitonic sort
// ---------------------------------------------------------------------------

typedef __bf16 bf16_t;
typedef bf16_t         v16bf   __attribute__((ext_vector_type(16)));
typedef unsigned short u16x16  __attribute__((ext_vector_type(16)));
typedef float          v8f     __attribute__((ext_vector_type(8)));
typedef unsigned int   uint32x4 __attribute__((ext_vector_type(4)));
typedef int            int32x4  __attribute__((ext_vector_type(4)));
typedef int            int32x8  __attribute__((ext_vector_type(8)));

__device__ __forceinline__ unsigned short bf_round(float x) {
    // round-to-nearest-even fp32 -> bf16 (bit pattern)
    unsigned u = __float_as_uint(x);
    u += 0x7FFFu + ((u >> 16) & 1u);
    return (unsigned short)(u >> 16);
}
__device__ __forceinline__ float bf_tof(unsigned short h) {
    return __uint_as_float(((unsigned)h) << 16);
}

extern __shared__ float smem[];   // TDM-staged tile: ICPG rows x PITCH f32 cols

// ---------------------------------------------------------------------------
// Generic grouped conv1d, C=256, 8 groups, 32 out-channels/group.
// Block: 256 threads = 8 waves; block covers 128 output columns of one
// 16-channel mtile; wave wv owns columns [wv*16, wv*16+16).
// TDM loads the block's activation tile (ICPG x (128+TAPS-1) f32) into LDS.
//   D(m,n) = sum_t sum_ic W[oc, ic, t] * X[ic, l0+n+t]   (cross-correlation)
// Reduction per tap in K=32 chunks via 3 bf16 WMMAs (hi*hi + lo*hi + hi*lo).
// SHUFFLE: channel-shuffle permutation on store.
// GATE:    out = gate[oc,l] * sigmoid(conv + bias)   (no shuffle in that mode)
// ---------------------------------------------------------------------------
template<int TAPS, int ICPG, bool SHUFFLE, bool GATE>
__global__ __launch_bounds__(256) void conv_wmma_kernel(
    const float* __restrict__ inA,   // primary input (sorted part for wa1)
    const float* __restrict__ inB,   // secondary input (h part for wa1 concat)
    const float* __restrict__ w,     // [256, ICPG, TAPS]
    const float* __restrict__ bias,  // [256]
    const float* __restrict__ gate,  // [8,256,Lout] or nullptr
    float* __restrict__ out,         // [8,256,Lout]
    int Lin, int Lout)
{
    constexpr int C     = 256;
    constexpr int PITCH = 128 + TAPS - 1;    // LDS tile width (elements)
    const int lane  = threadIdx.x & 31;
    const int wv    = threadIdx.x >> 5;      // wave id in block (8 waves)
    const int half  = lane >> 4;             // lane half (0/1)
    const int nl    = lane & 15;             // m for A-loads, n for B/D
    const int mtile = blockIdx.y;            // 0..15 -> 16 output channels each
    const int b     = blockIdx.z;            // batch
    const int gi    = mtile >> 1;            // group (32 oc per group)
    const int lb0   = blockIdx.x * 128;      // block's first output column
    const int l0    = lb0 + wv * 16;         // wave's first output column

    // ---- TDM: DMA the group's activation tile into LDS (wave 0 issues) ----
    if (wv == 0) {
        const float* src;
        int ch0;
        if (ICPG == 64) { src = (gi < 4) ? inA : inB; ch0 = (gi & 3) * 64; }
        else            { src = inA;                  ch0 = gi * 32;      }
        const uint64_t ga =
            (uint64_t)(uintptr_t)(src + ((size_t)(b * C + ch0) * (size_t)Lin + (size_t)lb0));
        const uint32_t remain = (uint32_t)(Lin - lb0);   // valid cols from tile start (OOB -> 0)

        uint32x4 g0;
        g0[0] = 1u;                                       // count=1, user descriptor
        g0[1] = 0u;                                       // lds_addr = 0 (dynamic LDS base)
        g0[2] = (uint32_t)(ga & 0xFFFFFFFFu);             // global_addr[31:0]
        g0[3] = (uint32_t)((ga >> 32) & 0x01FFFFFFu)      // global_addr[56:32]
              | (2u << 30);                               // type = 2 ("image")
        int32x8 g1;
        g1[0] = (int)(2u << 16);                          // data_size = 4 bytes
        g1[1] = (int)((remain & 0xFFFFu) << 16);          // tensor_dim0[15:0]
        g1[2] = (int)((remain >> 16) & 0xFFFFu)           // tensor_dim0[31:16]
              | (int)((uint32_t)ICPG << 16);              // tensor_dim1[15:0]
        g1[3] = (int)((uint32_t)PITCH << 16);             // tile_dim0
        g1[4] = (int)(uint32_t)ICPG;                      // tile_dim1 (tile_dim2 = 0)
        g1[5] = (int)(uint32_t)Lin;                       // tensor_dim0_stride[31:0]
        g1[6] = 0;                                        // stride hi / dim1 stride lo
        g1[7] = 0;
        int32x4 gz = {0, 0, 0, 0};
#if __clang_major__ >= 23
        int32x8 gz8 = {0, 0, 0, 0, 0, 0, 0, 0};
        __builtin_amdgcn_tensor_load_to_lds(g0, g1, gz, gz, gz8, 0);
#else
        __builtin_amdgcn_tensor_load_to_lds(g0, g1, gz, gz, 0);
#endif
        __builtin_amdgcn_s_wait_tensorcnt(0);
    }
    __syncthreads();

    if (l0 >= Lout) return;                  // wave-uniform exit

    v8f acc;
    #pragma unroll
    for (int r = 0; r < 8; ++r) acc[r] = 0.0f;

    constexpr int KC = ICPG / 32;            // K=32 chunks per tap
    #pragma unroll
    for (int t = 0; t < TAPS; ++t) {
        #pragma unroll
        for (int kc = 0; kc < KC; ++kc) {
            // ---- A operand: weights, 16x32 bf16 (hi/lo split) ----
            // lane holds row m = nl; element e -> k = (e&7) + 8*half + 16*(e>>3)
            u16x16 ahu, alu;
            const int oca = mtile * 16 + nl;
            #pragma unroll
            for (int e = 0; e < 16; ++e) {
                const int k   = (e & 7) + half * 8 + ((e >> 3) << 4);
                const int icg = kc * 32 + k;
                const float wval = w[(oca * ICPG + icg) * TAPS + t];
                const unsigned short h = bf_round(wval);
                ahu[e] = h;
                alu[e] = bf_round(wval - bf_tof(h));
            }
            // ---- B operand: activations from LDS tile, 32x16 bf16 ----
            // lane holds column n = nl; element e -> k = (e&7) + 8*half + 16*(e>>3)
            u16x16 bhu, blu;
            const int colbase = wv * 16 + t + nl;
            #pragma unroll
            for (int e = 0; e < 16; ++e) {
                const int row = kc * 32 + (e & 7) + half * 8 + ((e >> 3) << 4);
                const float xval = smem[row * PITCH + colbase];
                const unsigned short h = bf_round(xval);
                bhu[e] = h;
                blu[e] = bf_round(xval - bf_tof(h));
            }
            v16bf ah = __builtin_bit_cast(v16bf, ahu);
            v16bf al = __builtin_bit_cast(v16bf, alu);
            v16bf bh = __builtin_bit_cast(v16bf, bhu);
            v16bf bl = __builtin_bit_cast(v16bf, blu);
            // bf16x3 emulation of fp32 GEMM: hi*hi + lo*hi + hi*lo
            acc = __builtin_amdgcn_wmma_f32_16x16x32_bf16(false, ah, false, bh, (short)0, acc, false, false);
            acc = __builtin_amdgcn_wmma_f32_16x16x32_bf16(false, al, false, bh, (short)0, acc, false, false);
            acc = __builtin_amdgcn_wmma_f32_16x16x32_bf16(false, ah, false, bl, (short)0, acc, false, false);
        }
    }

    // ---- store D: vgpr r -> (m = r + 8*half, n = nl); 16-lane contiguous in L
    const int n = nl;
    #pragma unroll
    for (int r = 0; r < 8; ++r) {
        const int m  = r + half * 8;
        const int oc = mtile * 16 + m;
        if (l0 + n < Lout) {
            float v = acc[r] + bias[oc];
            if (GATE) {
                const float g = gate[(b * C + oc) * Lout + l0 + n];
                v = g * (1.0f / (1.0f + __expf(-v)));
            }
            const int chan = SHUFFLE ? ((oc & 31) * 8 + (oc >> 5)) : oc;
            out[(b * C + chan) * Lout + l0 + n] = v;
        }
    }
}

// ---------------------------------------------------------------------------
// Ascending sort of the 256 channels at one (b,l) position: LDS bitonic sort,
// one 256-thread workgroup per position (B*L = ~32.7K independent blocks).
// ---------------------------------------------------------------------------
__global__ __launch_bounds__(256) void sort_channels_kernel(
    const float* __restrict__ in, float* __restrict__ out, int Lc)
{
    __shared__ float s[256];
    const int pos = blockIdx.x;            // 0 .. B*Lc-1
    const int b   = pos / Lc;
    const int l   = pos - b * Lc;
    const int t   = threadIdx.x;
    s[t] = in[(b * 256 + t) * Lc + l];
    __syncthreads();
    for (int k = 2; k <= 256; k <<= 1) {
        for (int j = k >> 1; j > 0; j >>= 1) {
            const int ixj = t ^ j;
            if (ixj > t) {
                const float a = s[t];
                const float c = s[ixj];
                const bool asc = ((t & k) == 0);
                const bool swp = asc ? (a > c) : (a < c);
                if (swp) { s[t] = c; s[ixj] = a; }
            }
            __syncthreads();
        }
    }
    out[(b * 256 + t) * Lc + l] = s[t];
}

// ---------------------------------------------------------------------------

static inline int cdiv(int a, int b) { return (a + b - 1) / b; }

extern "C" void kernel_launch(void* const* d_in, const int* in_sizes, int n_in,
                              void* d_out, int out_size, void* d_ws, size_t ws_size,
                              hipStream_t stream) {
    (void)in_sizes; (void)n_in; (void)out_size; (void)ws_size;
    const float* x   = (const float*)d_in[0];
    const float* w1  = (const float*)d_in[1];
    const float* b1  = (const float*)d_in[2];
    const float* w2  = (const float*)d_in[3];
    const float* b2  = (const float*)d_in[4];
    const float* wa1 = (const float*)d_in[5];
    const float* ba1 = (const float*)d_in[6];
    const float* wa2 = (const float*)d_in[7];
    const float* ba2 = (const float*)d_in[8];
    const float* w3  = (const float*)d_in[9];
    const float* b3  = (const float*)d_in[10];
    float* out = (float*)d_out;

    constexpr int B = 8, C = 256;
    constexpr int L0 = 4096, L1 = 4094, L2 = 4092, L3 = 4088;
    constexpr size_t BUFELEMS = (size_t)B * C * 4096;   // padded slice
    float* buf0 = (float*)d_ws;                // h1 -> a1(shuffled)
    float* buf1 = buf0 + BUFELEMS;             // h2 (kept for attn + gate)
    float* buf2 = buf1 + BUFELEMS;             // sorted -> gated h

    const dim3 blk(256);
    auto grid_for = [](int Lout) { return dim3((unsigned)cdiv(Lout, 128), 16, 8); };
    auto shmem_for = [](int taps, int icpg) { return (size_t)icpg * (size_t)(128 + taps - 1) * 4; };

    // 1) h1 = shuffle(conv1(x))                     [B,C,L1]
    conv_wmma_kernel<3, 32, true, false><<<grid_for(L1), blk, shmem_for(3, 32), stream>>>(
        x, x, w1, b1, nullptr, buf0, L0, L1);
    // 2) h2 = shuffle(conv2(h1))                    [B,C,L2]
    conv_wmma_kernel<3, 32, true, false><<<grid_for(L2), blk, shmem_for(3, 32), stream>>>(
        buf0, buf0, w2, b2, nullptr, buf1, L1, L2);
    // 3) sorted = sort(h2, axis=channel)            [B,C,L2]
    sort_channels_kernel<<<dim3((unsigned)(B * L2)), blk, 0, stream>>>(buf1, buf2, L2);
    // 4) a1 = shuffle(conv(concat(sorted,h2), wa1)) [B,C,L2]; group<4 reads sorted, >=4 reads h2
    conv_wmma_kernel<1, 64, true, false><<<grid_for(L2), blk, shmem_for(1, 64), stream>>>(
        buf2, buf1, wa1, ba1, nullptr, buf0, L2, L2);
    // 5) hg = h2 * sigmoid(conv(a1, wa2))           [B,C,L2]
    conv_wmma_kernel<1, 32, false, true><<<grid_for(L2), blk, shmem_for(1, 32), stream>>>(
        buf0, buf0, wa2, ba2, buf1, buf2, L2, L2);
    // 6) y = conv3(hg)                              [B,C,L3]
    conv_wmma_kernel<5, 32, false, false><<<grid_for(L3), blk, shmem_for(5, 32), stream>>>(
        buf2, buf2, w3, b3, nullptr, out, L2, L3);
}
